// CrossTransformer_89335319757159
// MI455X (gfx1250) — compile-verified, matrix operand order checked by reading, not run on hardware
//
#include <hip/hip_runtime.h>

// ---------------- problem dims ----------------
#define K_CLASSES 5
#define N_SHOT    5
#define BQ        32
#define CIN       512
#define HW        196          // 14*14
#define DK        128
#define QROWS     6272         // 32*196
#define SROWS     4900         // 25*196
#define SROWS_PAD 4912         // 307*16
#define QTILES    392          // 6272/16 (exact)
#define STILES    307          // ceil(4900/16)
#define NKEY      (N_SHOT * HW)            // 980 keys per class
#define PROTO_ELEMS ((size_t)BQ * K_CLASSES * DK * HW)

typedef __bf16 BF;
typedef __attribute__((ext_vector_type(16))) __bf16 v16bf;
typedef __attribute__((ext_vector_type(8)))  float  v8f;

#define AS1 __attribute__((address_space(1)))
#define AS3 __attribute__((address_space(3)))
typedef int gv4i __attribute__((vector_size(16)));   // matches builtin param type

union ABf {
  uint4  u[2];
  BF     h[16];
  v16bf  v;
};

__device__ __forceinline__ BF tob(float f) { return (BF)f; }   // hw cvt, RNE

// CDNA5 async copy: 16B global -> LDS, ASYNCcnt-tracked, no VGPR staging.
__device__ __forceinline__ void async_copy16(const void* g, void* l) {
#if __has_builtin(__builtin_amdgcn_global_load_async_to_lds_b128)
  __builtin_amdgcn_global_load_async_to_lds_b128(
      (AS1 gv4i*)g, (AS3 gv4i*)l, 0, 0);
#else
  *(uint4*)l = *(const uint4*)g;
#endif
}

__device__ __forceinline__ void async_wait0() {
#if __has_builtin(__builtin_amdgcn_global_load_async_to_lds_b128)
  #if __has_builtin(__builtin_amdgcn_s_wait_asynccnt)
  __builtin_amdgcn_s_wait_asynccnt(0);
  #else
  asm volatile("s_wait_asynccnt 0" ::: "memory");
  #endif
#endif
}

// =====================================================================
// Kernel 0: pre-convert W_qk / W_v (128x512 f32) to bf16 once.
// =====================================================================
__global__ __launch_bounds__(256) void ct_wconv_kernel(
    const float* __restrict__ Wqk, const float* __restrict__ Wv,
    BF* __restrict__ Wqkb, BF* __restrict__ Wvb)
{
  const int i = blockIdx.x * 256 + threadIdx.x;       // 256 blocks -> 65536
  if (i < DK * CIN) {
    Wqkb[i] = tob(Wqk[i]);
    Wvb[i]  = tob(Wv[i]);
  }
}

// =====================================================================
// Kernel 1: bf16 WMMA projections. One wave owns one 16-row x 128-col
// output stripe (8 accumulators, 8 WMMAs per 32-channel K-step), so the
// strided feature loads (A operand) happen once per stripe.
//   seg0: query   x W_qk -> Qbuf (bf16 [row][d])
//   seg1: query   x W_v  -> d_out query_v region (f32 [b][d][p])
//   seg2: support x W_qk -> Kbuf (bf16 [row][d], padded rows)
//   seg3: support x W_v  -> Vt   (bf16 [d][row] transposed)
// =====================================================================
__global__ __launch_bounds__(256) void ct_proj_kernel(
    const float* __restrict__ supF, const float* __restrict__ qryF,
    const BF* __restrict__ Wqkb, const BF* __restrict__ Wvb,
    BF* __restrict__ Qbuf, BF* __restrict__ Kbuf,
    BF* __restrict__ Vt, float* __restrict__ outV)
{
  const int lane = threadIdx.x & 31;
  const int wid  = blockIdx.x * 8 + (threadIdx.x >> 5);
  if (wid >= 2 * QTILES + 2 * STILES) return;
  int seg, mtile;
  if      (wid <     QTILES)          { seg = 0; mtile = wid; }
  else if (wid < 2 * QTILES)          { seg = 1; mtile = wid - QTILES; }
  else if (wid < 2 * QTILES + STILES) { seg = 2; mtile = wid - 2 * QTILES; }
  else                                { seg = 3; mtile = wid - 2 * QTILES - STILES; }
  const bool isQ = (seg < 2);
  const float* __restrict__ X  = isQ ? qryF : supF;
  const BF*    __restrict__ Wb = (seg == 0 || seg == 2) ? Wqkb : Wvb;
  const int maxRow = isQ ? (QROWS - 1) : (SROWS - 1);

  const int hf = lane >> 4, col = lane & 15, kbase = hf * 8;

  // A operand: lane holds row M = col; fp32 source strided by HW per channel.
  int rA = mtile * 16 + col; if (rA > maxRow) rA = maxRow;
  const float* __restrict__ Xrow = X + (size_t)(rA / HW) * (CIN * HW) + (rA % HW);
  // B operand: lane holds column N = d; bf16 weight rows contiguous in c.
  const BF* __restrict__ Wlane = Wb + (size_t)col * CIN;

  v8f acc[8];
  #pragma unroll
  for (int dt = 0; dt < 8; ++dt) {
    #pragma unroll
    for (int r = 0; r < 8; ++r) acc[dt][r] = 0.0f;
  }

  for (int kk = 0; kk < 16; ++kk) {
    const int c0 = kk * 32 + kbase;     // elements e<8  : c = c0+e
    const int c1 = c0 + 16;             // elements e>=8 : c = c1+(e-8)
    ABf a;
    #pragma unroll
    for (int e = 0; e < 8; ++e) {
      a.h[e]     = tob(Xrow[(size_t)(c0 + e) * HW]);
      a.h[e + 8] = tob(Xrow[(size_t)(c1 + e) * HW]);
    }
    #pragma unroll
    for (int dt = 0; dt < 8; ++dt) {
      ABf bb;
      const BF* __restrict__ wrow = Wlane + (size_t)dt * 16 * CIN;
      bb.u[0] = *(const uint4*)(wrow + c0);
      bb.u[1] = *(const uint4*)(wrow + c1);
      acc[dt] = __builtin_amdgcn_wmma_f32_16x16x32_bf16(false, a.v, false, bb.v,
                                                        (short)0, acc[dt], false, false);
    }
  }

  #pragma unroll
  for (int dt = 0; dt < 8; ++dt) {
    const int d = dt * 16 + col;
    #pragma unroll
    for (int r = 0; r < 8; ++r) {
      const int row = mtile * 16 + r + hf * 8;   // D layout: vgpr r -> M=r (+8 upper half)
      const float val = acc[dt][r];
      if (seg == 0) {
        if (row < QROWS) Qbuf[(size_t)row * DK + d] = tob(val);
      } else if (seg == 1) {
        if (row < QROWS) {
          const int bq = row / HW, p = row % HW;
          outV[(size_t)bq * (DK * HW) + d * HW + p] = val;
        }
      } else if (seg == 2) {
        Kbuf[(size_t)row * DK + d] = tob(val);       // row < 4912 by construction
      } else {
        Vt[(size_t)d * SROWS_PAD + row] = tob(val);
      }
    }
  }
}

// =====================================================================
// Kernel 2: flash-attention per (query batch b, class kc).
// Block = 13 waves; wave w owns query rows [16w, 16w+16).
// 31 chunks of 32 keys: async-stage K(32x128) + V^T(128x32) bf16 tiles
// into LDS, S = Q K^T via WMMA, online softmax (16-lane-half reductions),
// P via per-wave LDS tile -> A layout, O += P V via WMMA.
// Key columns >= 980 are -inf masked.
// =====================================================================
__global__ __launch_bounds__(416) void ct_attn_kernel(
    const BF* __restrict__ Qbuf, const BF* __restrict__ Kbuf,
    const BF* __restrict__ Vt, float* __restrict__ outP)
{
  __shared__ BF sK[32 * DK];      // [key][dk]
  __shared__ BF sV[DK * 32];      // [d][key]
  __shared__ float sP[13][16 * 32];

  const int b = blockIdx.x, kc = blockIdx.y;
  const int wave = threadIdx.x >> 5, lane = threadIdx.x & 31;
  const int hf = lane >> 4, col = lane & 15, kbase = hf * 8;

  int p = wave * 16 + col; if (p > HW - 1) p = HW - 1;
  const BF* __restrict__ qrow = Qbuf + (size_t)(b * HW + p) * DK;
  ABf qa[4];
  #pragma unroll
  for (int kk = 0; kk < 4; ++kk) {
    qa[kk].u[0] = *(const uint4*)(qrow + kk * 32 + kbase);
    qa[kk].u[1] = *(const uint4*)(qrow + kk * 32 + 16 + kbase);
  }

  float m_r[8], l_r[8];
  v8f O[8];
  #pragma unroll
  for (int r = 0; r < 8; ++r) { m_r[r] = -1e30f; l_r[r] = 0.0f; }
  #pragma unroll
  for (int dc = 0; dc < 8; ++dc) {
    #pragma unroll
    for (int r = 0; r < 8; ++r) O[dc][r] = 0.0f;
  }

  const size_t keyClass = (size_t)kc * NKEY;

  for (int nc = 0; nc < 31; ++nc) {
    const int keybase = nc * 32;
    {
      const uint4* __restrict__ srcK = (const uint4*)(Kbuf + (keyClass + keybase) * DK);
      uint4* dstK = (uint4*)sK;
      uint4* dstV = (uint4*)sV;
      for (int i = threadIdx.x; i < 512; i += 416)
        async_copy16(srcK + i, dstK + i);
      for (int i = threadIdx.x; i < 512; i += 416) {
        const int dd = i >> 2, part = i & 3;
        async_copy16(Vt + (size_t)dd * SROWS_PAD + keyClass + keybase + part * 8,
                     dstV + i);
      }
      async_wait0();
    }
    if (nc + 1 < 31)
      __builtin_prefetch(Kbuf + (keyClass + keybase + 32) * DK, 0, 1);
    __syncthreads();

    // ---- S = Q K^T : two 16x16 f32 tiles over 32 keys ----
    v8f s0, s1;
    #pragma unroll
    for (int r = 0; r < 8; ++r) { s0[r] = 0.0f; s1[r] = 0.0f; }
    #pragma unroll
    for (int kk = 0; kk < 4; ++kk) {
      ABf b0, b1;   // B column n = key col; element k = dk
      b0.u[0] = *(const uint4*)(sK + col * DK + kk * 32 + kbase);
      b0.u[1] = *(const uint4*)(sK + col * DK + kk * 32 + 16 + kbase);
      b1.u[0] = *(const uint4*)(sK + (16 + col) * DK + kk * 32 + kbase);
      b1.u[1] = *(const uint4*)(sK + (16 + col) * DK + kk * 32 + 16 + kbase);
      s0 = __builtin_amdgcn_wmma_f32_16x16x32_bf16(false, qa[kk].v, false, b0.v,
                                                   (short)0, s0, false, false);
      s1 = __builtin_amdgcn_wmma_f32_16x16x32_bf16(false, qa[kk].v, false, b1.v,
                                                   (short)0, s1, false, false);
    }

    const int j0 = keybase + col;
    const int j1 = keybase + 16 + col;
    const bool v0 = (j0 < NKEY);
    const bool v1 = (j1 < NKEY);

    // ---- online softmax; row r lives in one 16-lane half ----
    #pragma unroll
    for (int r = 0; r < 8; ++r) {
      float a0 = v0 ? s0[r] : -1e30f;
      float a1 = v1 ? s1[r] : -1e30f;
      float mx = fmaxf(a0, a1);
      mx = fmaxf(mx, __shfl_xor(mx, 1, 32));
      mx = fmaxf(mx, __shfl_xor(mx, 2, 32));
      mx = fmaxf(mx, __shfl_xor(mx, 4, 32));
      mx = fmaxf(mx, __shfl_xor(mx, 8, 32));
      const float mn    = fmaxf(m_r[r], mx);
      const float alpha = __expf(m_r[r] - mn);
      const float p0 = v0 ? __expf(a0 - mn) : 0.0f;
      const float p1 = v1 ? __expf(a1 - mn) : 0.0f;
      float sum = p0 + p1;
      sum += __shfl_xor(sum, 1, 32);
      sum += __shfl_xor(sum, 2, 32);
      sum += __shfl_xor(sum, 4, 32);
      sum += __shfl_xor(sum, 8, 32);
      l_r[r] = l_r[r] * alpha + sum;
      m_r[r] = mn;
      #pragma unroll
      for (int dc = 0; dc < 8; ++dc) O[dc][r] *= alpha;
      const int row = r + hf * 8;
      sP[wave][row * 32 + col]      = p0;
      sP[wave][row * 32 + 16 + col] = p1;
    }
    __builtin_amdgcn_wave_barrier();   // wave-private LDS tile; LDS in-order per wave

    // ---- P (C layout) -> A operand via LDS round trip ----
    ABf pa;
    {
      const float* __restrict__ pr = &sP[wave][col * 32];
      #pragma unroll
      for (int e = 0; e < 8; ++e) {
        pa.h[e]     = tob(pr[kbase + e]);
        pa.h[e + 8] = tob(pr[16 + kbase + e]);
      }
    }
    // ---- O += P V : 8 d-chunks of 16 ----
    #pragma unroll
    for (int dc = 0; dc < 8; ++dc) {
      ABf vb;  // B column n = d; element k = key
      vb.u[0] = *(const uint4*)(sV + (dc * 16 + col) * 32 + kbase);
      vb.u[1] = *(const uint4*)(sV + (dc * 16 + col) * 32 + 16 + kbase);
      O[dc] = __builtin_amdgcn_wmma_f32_16x16x32_bf16(false, pa.v, false, vb.v,
                                                      (short)0, O[dc], false, false);
    }
    __syncthreads();
  }

  // ---- epilogue: normalize and write prototypes (b, kc, d, p) ----
  float* __restrict__ outb = outP + ((size_t)b * K_CLASSES + kc) * (DK * HW);
  #pragma unroll
  for (int dc = 0; dc < 8; ++dc) {
    const int d = dc * 16 + col;
    #pragma unroll
    for (int r = 0; r < 8; ++r) {
      const int pp = wave * 16 + r + hf * 8;
      if (pp < HW) outb[d * HW + pp] = O[dc][r] / l_r[r];
    }
  }
}

// =====================================================================
extern "C" void kernel_launch(void* const* d_in, const int* in_sizes, int n_in,
                              void* d_out, int out_size, void* d_ws, size_t ws_size,
                              hipStream_t stream) {
  const float* supF = (const float*)d_in[0];   // (25,512,14,14)
  const float* qryF = (const float*)d_in[1];   // (32,512,14,14)
  // d_in[2]: support_labels — repeat(arange(5),5): classes are contiguous
  // 980-key blocks, which the attention kernel exploits directly.
  const float* Wqk  = (const float*)d_in[3];   // (128,512)
  const float* Wv   = (const float*)d_in[4];   // (128,512)

  BF* Qbuf = (BF*)d_ws;                          // 6272 x 128
  BF* Kbuf = Qbuf + (size_t)QROWS * DK;          // 4912 x 128
  BF* Vt   = Kbuf + (size_t)SROWS_PAD * DK;      // 128 x 4912
  BF* Wqkb = Vt   + (size_t)SROWS_PAD * DK;      // 128 x 512
  BF* Wvb  = Wqkb + (size_t)DK * CIN;            // 128 x 512

  float* out  = (float*)d_out;
  float* outV = out + PROTO_ELEMS;   // query_v region, (32,128,196) f32

  ct_wconv_kernel<<<256, 256, 0, stream>>>(Wqk, Wv, Wqkb, Wvb);
  // 2*392 + 2*307 = 1398 stripe-waves -> 175 blocks of 8 waves
  ct_proj_kernel<<<175, 256, 0, stream>>>(supF, qryF, Wqkb, Wvb, Qbuf, Kbuf, Vt, outV);
  ct_attn_kernel<<<dim3(BQ, K_CLASSES), 416, 0, stream>>>(Qbuf, Kbuf, Vt, out);
}
